// PhasePicker_575525618143
// MI455X (gfx1250) — matrix operand order repaired and checked by reading, not compiled, not run on hardware
//
#include <hip/hip_runtime.h>
#include <math.h>

typedef unsigned short u16;
typedef __attribute__((ext_vector_type(16))) __bf16 v16bf;
typedef __attribute__((ext_vector_type(16))) unsigned short v16u;
typedef __attribute__((ext_vector_type(8)))  unsigned short v8u;
typedef __attribute__((ext_vector_type(8)))  float v8f;

__device__ __forceinline__ u16 f2bf(float f) {
    unsigned u = __builtin_bit_cast(unsigned, f);
    u += 0x7FFFu + ((u >> 16) & 1u);           // round-to-nearest-even
    return (u16)(u >> 16);
}
__device__ __forceinline__ float bf2f(u16 h) {
    return __builtin_bit_cast(float, (unsigned)h << 16);
}

// ---------------- f32 -> bf16 convert ----------------
__global__ void f32_to_bf16_kernel(const float* __restrict__ in, u16* __restrict__ out, long n) {
    long i = (long)blockIdx.x * blockDim.x + threadIdx.x;
    if (i < n) out[i] = f2bf(in[i]);
}

// ---------------- weight pre-pack into WMMA A-fragment layout ----------------
// K-ordering is k-major over padded channels: K = k*CP + ci, KTOT = 5*CP.
// A order used here (contiguous K, same as B): lane<16 -> K in {0..15}; lane>=16 -> K in {16..31}; M = lane%16.
// BN scale folded into the packed weights.
__global__ void prepack_kernel(const float* __restrict__ w, const float* __restrict__ g,
                               const float* __restrict__ bb, const float* __restrict__ m,
                               const float* __restrict__ v, int CIN, int CP, int COUT, int hasBN,
                               u16* __restrict__ Apack, float* __restrict__ shift, int total) {
    int idx = blockIdx.x * blockDim.x + threadIdx.x;
    if (idx < COUT) {
        float sc = hasBN ? g[idx] * rsqrtf(v[idx] + 1e-5f) : 1.0f;
        shift[idx] = hasBN ? (bb[idx] - m[idx] * sc) : bb[idx];
    }
    if (idx >= total) return;
    const int KTOT = CP * 5;
    const int KC = (KTOT + 31) / 32;
    int e = idx & 15;
    int lane = (idx >> 4) & 31;
    int kc = (idx >> 9) % KC;
    int cot = (idx >> 9) / KC;
    int co = cot * 16 + (lane & 15);
    int K = kc * 32 + ((lane < 16) ? 0 : 16) + e;
    float val = 0.0f;
    if (co < COUT && K < KTOT) {
        int k = K / CP, ci = K % CP;
        if (ci < CIN) {
            float sc = hasBN ? g[co] * rsqrtf(v[co] + 1e-5f) : 1.0f;
            val = w[(co * CIN + ci) * 5 + k] * sc;
        }
    }
    Apack[idx] = f2bf(val);
}

// ---------------- fused reflect-pad dilated conv1d + BN + ReLU (WMMA bf16) ----------------
// in : bf16 [256][CIN][LIN], out : bf16 [256][COUT][LIN]
// GEMM: M=COUT (16-row tile per wave), K=5*CP (k-major), N=64 positions per block.
// LDS window staged channel-contiguous: sx[pos][ci], row stride CPS -> B fragment
// is 16 contiguous bf16 = two ds_load_b128 per WMMA.
template <int CIN, int COUT, int DIL, int LIN>
__global__ __launch_bounds__((((COUT + 15) / 16) * 32))
void conv_bn_relu_wmma(const u16* __restrict__ in, const u16* __restrict__ Apack,
                       const float* __restrict__ shift, u16* __restrict__ out, int relu) {
    constexpr int CP   = (CIN < 16) ? 16 : CIN;   // padded channels (power of two)
    constexpr int KTOT = CP * 5;
    constexpr int KC   = (KTOT + 31) / 32;
    constexpr int COT  = (COUT + 15) / 16;
    constexpr int P    = 2 * DIL;                 // reflect pad = ((K-1)*d+1)//2
    constexpr int W    = 64 + 4 * DIL;            // staged window width
    constexpr int CPS  = CP + 8;                  // row stride (keeps 16B align, breaks bank conflicts)
    __shared__ u16 sx[W * CPS];

    const int n  = blockIdx.y;
    const int l0 = blockIdx.x * 64;

    // Stage reflect-padded bf16 window into LDS, transposed to [pos][ci].
    for (int idx = threadIdx.x; idx < CP * W; idx += COT * 32) {
        int ci = idx / W, ww = idx - ci * W;      // coalesced global reads per-ci row
        int gpos = l0 - P + ww;
        if (gpos < 0) gpos = -gpos;
        if (gpos >= LIN) gpos = 2 * LIN - 2 - gpos;
        u16 val = (ci < CIN) ? in[((size_t)n * CIN + ci) * LIN + gpos] : (u16)0;
        sx[ww * CPS + ci] = val;
    }
    __syncthreads();

    const int wave = threadIdx.x >> 5;
    const int lane = threadIdx.x & 31;
    const int mcol = lane & 15;                   // B/C column (N index)
    const int kbB  = (lane < 16) ? 0 : 16;        // B K-half per lane group

    v8f acc[4] = {};

    for (int kc = 0; kc < KC; ++kc) {
        // A fragment: one contiguous 32B load from pre-packed weights
        v16bf a = *reinterpret_cast<const v16bf*>(
            Apack + (((size_t)wave * KC + kc) * 32 + lane) * 16);
        const int K0 = kc * 32 + kbB;             // start of this lane's 16-run (multiple of 16)
        const int k   = K0 / CP;                  // CP is a power of two -> shifts
        const int ci0 = K0 % CP;
#pragma unroll
        for (int j = 0; j < 4; ++j) {
            v16bf b;
            if (K0 < KTOT) {
                int addr = (j * 16 + mcol + k * DIL) * CPS + ci0;  // 16B-aligned
                v8u lo = *reinterpret_cast<const v8u*>(&sx[addr]);
                v8u hi = *reinterpret_cast<const v8u*>(&sx[addr + 8]);
                b = __builtin_bit_cast(v16bf, __builtin_shufflevector(
                        lo, hi, 0, 1, 2, 3, 4, 5, 6, 7, 8, 9, 10, 11, 12, 13, 14, 15));
            } else {
                v16u z = {};
                b = __builtin_bit_cast(v16bf, z);
            }
            acc[j] = __builtin_amdgcn_wmma_f32_16x16x32_bf16(
                false, a, false, b, (short)0, acc[j], false, false);
        }
    }

    // Epilogue: + BN shift (or bias), optional ReLU, store bf16.
    const int mo = (lane < 16) ? 0 : 8;           // C layout M offset
#pragma unroll
    for (int j = 0; j < 4; ++j) {
#pragma unroll
        for (int i = 0; i < 8; ++i) {
            int co = wave * 16 + mo + i;
            if (co < COUT) {
                float vv = acc[j][i] + shift[co];
                if (relu) vv = fmaxf(vv, 0.0f);
                out[((size_t)n * COUT + co) * LIN + l0 + j * 16 + mcol] = f2bf(vv);
            }
        }
    }
}

// ---------------- 2x linear upsample (align_corners=False), bf16 -> bf16 ----------------
__global__ void up2_kernel(const u16* __restrict__ in, u16* __restrict__ out, int L, long total) {
    long idx = (long)blockIdx.x * blockDim.x + threadIdx.x;
    if (idx >= total) return;
    long row = idx / (2 * L);
    int j = (int)(idx - row * (2 * L));
    const u16* r = in + row * (long)L;
    int l = j >> 1;
    float val;
    if ((j & 1) == 0)
        val = (l == 0) ? bf2f(r[0]) : 0.25f * bf2f(r[l - 1]) + 0.75f * bf2f(r[l]);
    else
        val = (l == L - 1) ? bf2f(r[L - 1]) : 0.75f * bf2f(r[l]) + 0.25f * bf2f(r[l + 1]);
    out[idx] = f2bf(val);
}

// ---------------- final up2 + transpose to (b,3,t,s), f32 out ----------------
__global__ void final_up2_kernel(const u16* __restrict__ h5, float* __restrict__ out) {
    const long total = 32L * 3 * 32768 * 8;
    long idx = (long)blockIdx.x * blockDim.x + threadIdx.x;
    if (idx >= total) return;
    int s = idx & 7;
    long r = idx >> 3;
    int t = (int)(r & 32767);
    r >>= 15;                 // r = b*3 + c
    int c = (int)(r % 3);
    int b = (int)(r / 3);
    const int L = 16384;
    const u16* row = h5 + ((size_t)((b * 8 + s) * 3 + c)) * L;
    int l = t >> 1;
    float val;
    if ((t & 1) == 0)
        val = (l == 0) ? bf2f(row[0]) : 0.25f * bf2f(row[l - 1]) + 0.75f * bf2f(row[l]);
    else
        val = (l == L - 1) ? bf2f(row[L - 1]) : 0.75f * bf2f(row[l]) + 0.25f * bf2f(row[l + 1]);
    out[idx] = val;
}

// ---------------- loss pieces ----------------
__global__ void init_acc_kernel(float* acc) {
    if (threadIdx.x < 2) acc[threadIdx.x] = 0.0f;
}

// one block per (b,s): count pairs where targets[b,:,:,s] has any nonzero
__global__ void num_kernel(const float* __restrict__ targets, float* __restrict__ acc) {
    __shared__ int flag;
    if (threadIdx.x == 0) flag = 0;
    __syncthreads();
    int b = blockIdx.x >> 3, s = blockIdx.x & 7;
    bool found = false;
    for (int i = threadIdx.x; i < 3 * 32768; i += blockDim.x) {
        int c = i / 32768, t = i - c * 32768;
        if (targets[(((long)b * 3 + c) * 32768 + t) * 8 + s] != 0.0f) { found = true; break; }
    }
    if (found) flag = 1;
    __syncthreads();
    if (threadIdx.x == 0 && flag) atomicAdd(acc + 1, 1.0f);
}

__global__ void loss_kernel(const float* __restrict__ out, const float* __restrict__ targets,
                            float* __restrict__ acc) {
    const long total = 32L * 32768 * 8;
    long idx = (long)blockIdx.x * blockDim.x + threadIdx.x;
    float partial = 0.0f;
    if (idx < total) {
        int s = idx & 7;
        long r = idx >> 3;
        int t = (int)(r & 32767);
        int b = (int)(r >> 15);
        long base = (((long)b * 3) * 32768 + t) * 8 + s;
        const long stride = 32768L * 8;
        float o0 = out[base], o1 = out[base + stride], o2 = out[base + 2 * stride];
        float mx = fmaxf(o0, fmaxf(o1, o2));
        float lse = mx + logf(expf(o0 - mx) + expf(o1 - mx) + expf(o2 - mx));
        float t0 = targets[base], t1 = targets[base + stride], t2 = targets[base + 2 * stride];
        partial = -(t0 * (o0 - lse) + t1 * (o1 - lse) + t2 * (o2 - lse));
    }
    __shared__ float red[256];
    red[threadIdx.x] = partial;
    __syncthreads();
    for (int off = 128; off > 0; off >>= 1) {
        if (threadIdx.x < off) red[threadIdx.x] += red[threadIdx.x + off];
        __syncthreads();
    }
    if (threadIdx.x == 0) atomicAdd(acc, red[0]);
}

__global__ void finalize_kernel(const float* __restrict__ acc, float* __restrict__ out_loss) {
    *out_loss = acc[0] / (acc[1] * 32768.0f);
}

// ---------------- host launch ----------------
extern "C" void kernel_launch(void* const* d_in, const int* in_sizes, int n_in,
                              void* d_out, int out_size, void* d_ws, size_t ws_size,
                              hipStream_t stream) {
    const float* x       = (const float*)d_in[0];
    const float* targets = (const float*)d_in[1];
    const float* wl[4], *gl[4], *bl[4], *ml[4], *vl[4];
    for (int i = 0; i < 4; ++i) {
        wl[i] = (const float*)d_in[2 + 5 * i];
        gl[i] = (const float*)d_in[3 + 5 * i];
        bl[i] = (const float*)d_in[4 + 5 * i];
        ml[i] = (const float*)d_in[5 + 5 * i];
        vl[i] = (const float*)d_in[6 + 5 * i];
    }
    const float* w_out = (const float*)d_in[22];
    const float* b_out = (const float*)d_in[23];
    float* fout = (float*)d_out;

    // workspace layout: three 64 MiB ping-pong bf16 regions + packs + accumulators
    const size_t REG = 67108864ull;
    char* base = (char*)d_ws;
    u16* regA = (u16*)(base);
    u16* regB = (u16*)(base + REG);
    u16* regC = (u16*)(base + 2 * REG);
    char* p = base + 3 * REG;
    auto alloc = [&](size_t bytes) { void* r = (void*)p; p += (bytes + 255) & ~(size_t)255; return r; };
    u16* ap0 = (u16*)alloc((size_t)4 * 20 * 512 * 2);   // CP=128, KC=20, COT=4
    u16* ap1 = (u16*)alloc((size_t)2 * 10 * 512 * 2);   // CP=64,  KC=10, COT=2
    u16* ap2 = (u16*)alloc((size_t)1 * 5 * 512 * 2);    // CP=32,  KC=5
    u16* ap3 = (u16*)alloc((size_t)1 * 3 * 512 * 2);    // CP=16,  KC=3
    u16* ap4 = (u16*)alloc((size_t)1 * 3 * 512 * 2);    // CP=16,  KC=3 (CIN=8 padded)
    float* sh0 = (float*)alloc(256);
    float* sh1 = (float*)alloc(256);
    float* sh2 = (float*)alloc(256);
    float* sh3 = (float*)alloc(256);
    float* sh4 = (float*)alloc(256);
    float* acc = (float*)alloc(256);

    init_acc_kernel<<<1, 32, 0, stream>>>(acc);

    // x (256,128,1024) f32 -> bf16
    const long NX = 33554432L;
    f32_to_bf16_kernel<<<(unsigned)((NX + 255) / 256), 256, 0, stream>>>(x, regA, NX);

    // weight pre-pack (BN scale folded), shifts
    prepack_kernel<<<(4 * 20 * 512 + 255) / 256, 256, 0, stream>>>(wl[0], gl[0], bl[0], ml[0], vl[0], 128, 128, 64, 1, ap0, sh0, 4 * 20 * 512);
    prepack_kernel<<<(2 * 10 * 512 + 255) / 256, 256, 0, stream>>>(wl[1], gl[1], bl[1], ml[1], vl[1], 64, 64, 32, 1, ap1, sh1, 2 * 10 * 512);
    prepack_kernel<<<(1 * 5 * 512 + 255) / 256, 256, 0, stream>>>(wl[2], gl[2], bl[2], ml[2], vl[2], 32, 32, 16, 1, ap2, sh2, 1 * 5 * 512);
    prepack_kernel<<<(1 * 3 * 512 + 255) / 256, 256, 0, stream>>>(wl[3], gl[3], bl[3], ml[3], vl[3], 16, 16, 8, 1, ap3, sh3, 1 * 3 * 512);
    prepack_kernel<<<(1 * 3 * 512 + 255) / 256, 256, 0, stream>>>(w_out, b_out, b_out, b_out, b_out, 8, 16, 3, 0, ap4, sh4, 1 * 3 * 512);

    const long UP = 33554432L; // every upsample pass writes 33.5M elements
    const unsigned upg = (unsigned)((UP + 255) / 256);

    conv_bn_relu_wmma<128, 64, 1, 1024><<<dim3(16, 256), 128, 0, stream>>>(regA, ap0, sh0, regB, 1);
    up2_kernel<<<upg, 256, 0, stream>>>(regB, regC, 1024, UP);

    conv_bn_relu_wmma<64, 32, 2, 2048><<<dim3(32, 256), 64, 0, stream>>>(regC, ap1, sh1, regA, 1);
    up2_kernel<<<upg, 256, 0, stream>>>(regA, regB, 2048, UP);

    conv_bn_relu_wmma<32, 16, 4, 4096><<<dim3(64, 256), 32, 0, stream>>>(regB, ap2, sh2, regC, 1);
    up2_kernel<<<upg, 256, 0, stream>>>(regC, regA, 4096, UP);

    conv_bn_relu_wmma<16, 8, 8, 8192><<<dim3(128, 256), 32, 0, stream>>>(regA, ap3, sh3, regB, 1);
    up2_kernel<<<upg, 256, 0, stream>>>(regB, regC, 8192, UP);

    conv_bn_relu_wmma<8, 3, 16, 16384><<<dim3(256, 256), 32, 0, stream>>>(regC, ap4, sh4, regA, 0);

    const long NOUT = 25165824L;
    final_up2_kernel<<<(unsigned)((NOUT + 255) / 256), 256, 0, stream>>>(regA, fout);

    num_kernel<<<256, 256, 0, stream>>>(targets, acc);
    loss_kernel<<<32768, 256, 0, stream>>>(fout, targets, acc);
    finalize_kernel<<<1, 1, 0, stream>>>(acc, fout + NOUT);
}
